// HAU_10892037062772
// MI455X (gfx1250) — compile-verified, matrix operand order checked by reading, not run on hardware
//
#include <hip/hip_runtime.h>
#include <hip/hip_bf16.h>
#include <math.h>

typedef __attribute__((ext_vector_type(16))) _Float16 v16h;
typedef __attribute__((ext_vector_type(8)))  _Float16 v8h;
typedef __attribute__((ext_vector_type(4)))  _Float16 v4h;
typedef __attribute__((ext_vector_type(8)))  float    v8f;

#define BATCH 8
#define HW    4096   // 64*64

// ---------------- workspace layout (bytes) ----------------
#define OFF_XH    (size_t)0                      // f16  8*256*4096        = 16 MB
#define OFF_COMP  (size_t)16777216               // f16  8*128*4096        =  8 MB
#define OFF_T     (size_t)(16777216+8388608)     // f16  8*64*4096         =  4 MB
#define OFF_KW    (OFF_T + (size_t)4194304)      // f32  8*4096*36         =  4.5 MB
#define OFF_ENH   (OFF_KW + (size_t)4718592)     // f32  8*256*4096        = 32 MB
#define OFF_POOL  (OFF_ENH + (size_t)33554432)   // f32  8*256
#define OFF_GATE  (OFF_POOL + (size_t)8192)      // f32  8*256

// ---------------- WMMA fragment helpers (CDNA5 16x16x32 f16 layout) ----------------
// A tile in LDS: row-major [16][ld]; per-lane reads are 2x16B contiguous -> ds_load_b128.
__device__ __forceinline__ v16h load_frag_A(const _Float16* p, int ld) {
  int lane = threadIdx.x & 31;
  int r  = lane & 15;
  int kb = (lane >> 4) << 3;          // lanes 0-15: K {0..7,16..23}; 16-31: {8..15,24..31}
  v16h a;
#pragma unroll
  for (int e = 0; e < 16; ++e) {
    int k = (e < 8) ? (kb + e) : (kb + 8 + e);
    a[e] = p[r * ld + k];
  }
  return a;
}
// B tile stored pre-swizzled in fragment order: [nt][lane][16 halves] (32B/lane contiguous)
__device__ __forceinline__ v16h load_frag_B_sw(const _Float16* p) {
  int lane = threadIdx.x & 31;
  return *(const v16h*)(p + lane * 16);
}

// ---------------- 0: f32 -> f16 cast of x (float4 -> packed 4xf16) ----------------
__global__ void cast_f16_kernel(const float* __restrict__ x, _Float16* __restrict__ xh) {
  size_t i = (size_t)blockIdx.x * 256 + threadIdx.x;
  float4 v = ((const float4*)x)[i];
  v4h h;
  h[0] = (_Float16)v.x; h[1] = (_Float16)v.y; h[2] = (_Float16)v.z; h[3] = (_Float16)v.w;
  ((v4h*)xh)[i] = h;
}

// ---------------- 1: comp = 1x1 conv (WMMA GEMM 128x4096x256 per batch) ----------------
// block tile 128(M) x 32(N): 8 waves = 8 M-tiles, each wave 2 N-tiles. Double-buffered.
// All LDS staging is done in 16B fragment chunks -> ds_store_b128.
__global__ void comp_wmma_kernel(const _Float16* __restrict__ xh, const float* __restrict__ Wkc,
                                 const float* __restrict__ bkc, _Float16* __restrict__ comp_h) {
  __shared__ alignas(32) _Float16 As[2][128 * 32];
  __shared__ alignas(32) _Float16 Bs[2][32 * 32];   // swizzled, 2 N-subtiles
  int tid = threadIdx.x;
  int b   = blockIdx.y;
  int n0  = blockIdx.x * 32;
  const _Float16* xb = xh + (size_t)b * 256 * HW;

  auto stage = [&](int buf, int kt) {
    // A: 128x32 halves = 512 chunks of 8; 2 chunks/thread; rows contiguous
#pragma unroll
    for (int i = 0; i < 2; ++i) {
      int ch = tid + i * 256;
      int m  = ch >> 2;
      int c0 = (ch & 3) * 8;
      const float* wp = &Wkc[m * 256 + kt + c0];
      v8h pk;
#pragma unroll
      for (int j = 0; j < 8; ++j) pk[j] = (_Float16)wp[j];
      *(v8h*)&As[buf][m * 32 + c0] = pk;
    }
    // B: 32x32 halves = 128 chunks of 8 (8 consecutive K, fixed column), swizzled
    if (tid < 128) {
      int half = tid & 1;
      int lane = (tid >> 1) & 31;
      int nt   = tid >> 6;                    // 0..1
      int n    = n0 + nt * 16 + (lane & 15);
      int kb   = (lane >> 4) << 3;
      int k0   = kt + (half ? kb + 16 : kb);
      v8h pk;
#pragma unroll
      for (int j = 0; j < 8; ++j) pk[j] = xb[(size_t)(k0 + j) * HW + n];
      *(v8h*)&Bs[buf][((nt * 32 + lane) * 2 + half) * 8] = pk;
    }
  };

  v8f accs[2] = {};
  int wv = tid >> 5;                        // M tile id
  stage(0, 0);
  __syncthreads();
#pragma unroll
  for (int it = 0; it < 8; ++it) {          // K = 256, chunks of 32, fully unrolled
    int buf = it & 1;
    if (it < 7) stage(buf ^ 1, (it + 1) * 32);
    v16h a  = load_frag_A(As[buf] + wv * 16 * 32, 32);
    v16h b0 = load_frag_B_sw(Bs[buf]);
    v16h b1 = load_frag_B_sw(Bs[buf] + 32 * 16);
    accs[0] = __builtin_amdgcn_wmma_f32_16x16x32_f16(false, a, false, b0, (short)0, accs[0], false, false);
    accs[1] = __builtin_amdgcn_wmma_f32_16x16x32_f16(false, a, false, b1, (short)0, accs[1], false, false);
    __syncthreads();
  }
  int lane = tid & 31;
  int col = lane & 15, rb = (lane >> 4) * 8;
  _Float16* op = comp_h + (size_t)b * 128 * HW;
#pragma unroll
  for (int j = 0; j < 2; ++j)
#pragma unroll
    for (int i = 0; i < 8; ++i) {
      int m = wv * 16 + rb + i;
      int n = n0 + j * 16 + col;
      float v = accs[j][i] + bkc[m];
      op[(size_t)m * HW + n] = (_Float16)v;
    }
}

// ---------------- 2/5: 3x3 conv via implicit im2col WMMA GEMM ----------------
// block tile 64(M) x 64(N): 8 waves = 4 M-tiles x 2 N-quads, 2 accumulators/wave.
// MODE 0: +bias, BN, SiLU, f16 out (kernel predictor).  MODE 1: raw f32 out (grouped conv).
template <int CIN, int NG, int MODE>
__global__ void conv3x3_wmma_kernel(const _Float16* __restrict__ in, const float* __restrict__ W,
                                    const float* __restrict__ bias,
                                    const float* __restrict__ bn_g, const float* __restrict__ bn_b,
                                    const float* __restrict__ bn_m, const float* __restrict__ bn_v,
                                    _Float16* __restrict__ out_h, float* __restrict__ out_f) {
  constexpr int KTOT = CIN * 9;
  constexpr int NCHUNK = KTOT / 32;          // 36 (CIN=128) or 18 (CIN=64)
  __shared__ alignas(32) _Float16 As[2][64 * 32];
  __shared__ alignas(32) _Float16 Bs[2][64 * 32];   // swizzled: [nt(4)][lane(32)][16]
  int tid = threadIdx.x;
  int gy  = blockIdx.y;                      // = b*NG + g
  int g   = gy % NG;
  size_t inBase  = (size_t)gy * CIN * HW;    // contiguous group slices
  size_t outBase = (size_t)gy * 64 * HW;
  size_t wBase   = (size_t)g * 64 * KTOT;
  int n0 = blockIdx.x * 64;

  auto stage = [&](int buf, int kt) {
    // A: 64x32 halves = 256 chunks of 8; 1 chunk/thread; rows contiguous
    {
      int m  = tid >> 2;
      int c0 = (tid & 3) * 8;
      const float* wp = &W[wBase + (size_t)m * KTOT + kt + c0];
      v8h pk;
#pragma unroll
      for (int j = 0; j < 8; ++j) pk[j] = (_Float16)wp[j];
      *(v8h*)&As[buf][m * 32 + c0] = pk;
    }
    if (kt + 32 < KTOT)                      // prefetch next weight chunk -> global_prefetch
      __builtin_prefetch(&W[wBase + (size_t)(tid >> 2) * KTOT + kt + 32], 0, 1);
    // B: 32K x 64N halves = 256 chunks of 8 consecutive-K (implicit im2col, zero pad=1)
    {
      int half = tid & 1;
      int lane = (tid >> 1) & 31;
      int nt   = tid >> 6;                   // 0..3
      int n    = n0 + nt * 16 + (lane & 15);
      int kb   = (lane >> 4) << 3;
      int k0   = kt + (half ? kb + 16 : kb);
      int yb = n >> 6, xb = n & 63;
      v8h pk;
#pragma unroll
      for (int j = 0; j < 8; ++j) {
        int k  = k0 + j;
        int ci = k / 9;
        int tap = k - ci * 9;
        int t3  = tap / 3;
        int y = yb + t3 - 1, x = xb + (tap - t3 * 3) - 1;
        _Float16 v = (_Float16)0.f;
        if ((unsigned)y < 64u && (unsigned)x < 64u)
          v = in[inBase + (size_t)ci * HW + y * 64 + x];
        pk[j] = v;
      }
      *(v8h*)&Bs[buf][((nt * 32 + lane) * 2 + half) * 8] = pk;
    }
  };

  v8f accs[2] = {};
  int wv = tid >> 5;
  int mt = wv & 3, nq = wv >> 2;

  stage(0, 0);
  __syncthreads();
#pragma unroll 2
  for (int it = 0; it < NCHUNK; ++it) {
    int buf = it & 1;
    if (it + 1 < NCHUNK) stage(buf ^ 1, (it + 1) * 32);
    v16h a  = load_frag_A(As[buf] + mt * 16 * 32, 32);
    v16h b0 = load_frag_B_sw(Bs[buf] + (nq * 2 + 0) * 32 * 16);
    v16h b1 = load_frag_B_sw(Bs[buf] + (nq * 2 + 1) * 32 * 16);
    accs[0] = __builtin_amdgcn_wmma_f32_16x16x32_f16(false, a, false, b0, (short)0, accs[0], false, false);
    accs[1] = __builtin_amdgcn_wmma_f32_16x16x32_f16(false, a, false, b1, (short)0, accs[1], false, false);
    __syncthreads();
  }

  int lane = tid & 31;
  int col = lane & 15, rb = (lane >> 4) * 8;
#pragma unroll
  for (int j = 0; j < 2; ++j)
#pragma unroll
    for (int i = 0; i < 8; ++i) {
      int m = mt * 16 + rb + i;
      int n = n0 + (nq * 2 + j) * 16 + col;
      float v = accs[j][i];
      if constexpr (MODE == 0) {
        v += bias[m];
        float sc = bn_g[m] * rsqrtf(bn_v[m] + 1e-5f);
        v = (v - bn_m[m]) * sc + bn_b[m];
        v = v / (1.f + __expf(-v));          // SiLU
        out_h[outBase + (size_t)m * HW + n] = (_Float16)v;
      } else {
        out_f[outBase + (size_t)m * HW + n] = v;
      }
    }
}

// ---------------- 3: kp2 1x1 conv (64 -> 36) + per-group softmax, zero pooled ----------------
__global__ void kp2_softmax_kernel(const _Float16* __restrict__ t_h, const float* __restrict__ Wkp2,
                                   const float* __restrict__ bkp2, float* __restrict__ kw,
                                   float* __restrict__ pooled) {
  __shared__ float Ws[36 * 64];
  __shared__ float bS[36];
  int tid = threadIdx.x;
  for (int i = tid; i < 36 * 64; i += 256) Ws[i] = Wkp2[i];
  if (tid < 36) bS[tid] = bkp2[tid];
  if (blockIdx.x < 8) pooled[blockIdx.x * 256 + tid] = 0.f;   // zero SE accumulator
  __syncthreads();
  int n  = blockIdx.x * 256 + tid;
  int b  = n >> 12;
  int hw = n & 4095;
  float tv[64];
  const _Float16* tp = t_h + (size_t)b * 64 * HW + hw;
#pragma unroll
  for (int ci = 0; ci < 64; ++ci) tv[ci] = (float)tp[(size_t)ci * HW];
  float o[36];
#pragma unroll 4
  for (int oo = 0; oo < 36; ++oo) {
    float a = bS[oo];
#pragma unroll
    for (int ci = 0; ci < 64; ++ci) a += Ws[oo * 64 + ci] * tv[ci];
    o[oo] = a;
  }
  float* kp = kw + (size_t)n * 36;
#pragma unroll
  for (int gg = 0; gg < 4; ++gg) {                    // softmax over each k^2=9 group
    float mx = o[gg * 9];
#pragma unroll
    for (int k = 1; k < 9; ++k) mx = fmaxf(mx, o[gg * 9 + k]);
    float e[9], s = 0.f;
#pragma unroll
    for (int k = 0; k < 9; ++k) { e[k] = __expf(o[gg * 9 + k] - mx); s += e[k]; }
    float inv = 1.f / s;
#pragma unroll
    for (int k = 0; k < 9; ++k) kp[gg * 9 + k] = e[k] * inv;
  }
}

// ---------------- 6: reassembly + bilinear residual + fused write + pooled partial ----------------
// block (64 w-pixels, 4 channels); grid (b*64 rows, 64 channel-groups)
__global__ void reassembly_kernel(const float* __restrict__ x, const float* __restrict__ kw,
                                  const float* __restrict__ enh, float* __restrict__ out,
                                  float* __restrict__ pooled) {
  __shared__ float kwS[64 * 36];
  __shared__ float red[256];
  int tx = threadIdx.x, ty = threadIdx.y;
  int lin = ty * 64 + tx;
  int b = blockIdx.x >> 6;
  int h = blockIdx.x & 63;
  int c = blockIdx.y * 4 + ty;
  {
    const float* kr = kw + ((size_t)(b * HW) + h * 64) * 36;
#pragma unroll
    for (int i = 0; i < 9; ++i) kwS[lin + i * 256] = kr[lin + i * 256];
  }
  __syncthreads();
  // 9-tap patch of x at (h, tx), zero pad
  const float* xp = x + (size_t)(b * 256 + c) * HW;
  float xv[9];
#pragma unroll
  for (int dy = -1; dy <= 1; ++dy)
#pragma unroll
    for (int dx = -1; dx <= 1; ++dx) {
      int yy = h + dy, xx = tx + dx;
      xv[(dy + 1) * 3 + dx + 1] =
          ((unsigned)yy < 64u && (unsigned)xx < 64u) ? xp[yy * 64 + xx] : 0.f;
    }
  float acc[4] = {0.f, 0.f, 0.f, 0.f};
  const float* kwp = &kwS[tx * 36];
#pragma unroll
  for (int s = 0; s < 4; ++s)
#pragma unroll
    for (int k = 0; k < 9; ++k) acc[s] += xv[k] * kwp[s * 9 + k];
  // reference reshape mapping: s2 -> (ys = 2h + s2/2, xs = (s2&1)*64 + tx)
  const float* ep = enh + (size_t)(b * 256 + c) * HW;
  float psum = 0.f;
#pragma unroll
  for (int s = 0; s < 4; ++s) {
    int ys = 2 * h + (s >> 1);
    int xs = ((s & 1) << 6) + tx;
    float sy = ys * 0.5f - 0.25f;
    float sx = xs * 0.5f - 0.25f;
    int iy0 = (int)floorf(sy), ix0 = (int)floorf(sx);
    float fy = sy - (float)iy0, fx = sx - (float)ix0;
    int y0 = iy0 < 0 ? 0 : iy0, y1 = iy0 + 1 > 63 ? 63 : iy0 + 1;
    int x0 = ix0 < 0 ? 0 : ix0, x1 = ix0 + 1 > 63 ? 63 : ix0 + 1;
    float up = (1.f - fy) * ((1.f - fx) * ep[y0 * 64 + x0] + fx * ep[y0 * 64 + x1]) +
               fy * ((1.f - fx) * ep[y1 * 64 + x0] + fx * ep[y1 * 64 + x1]);
    float v = acc[s] + up;
    out[(size_t)(b * 256 + c) * 16384 + ys * 128 + xs] = v;
    psum += v;
  }
  red[lin] = psum;
  __syncthreads();
  for (int off = 32; off > 0; off >>= 1) {
    if (tx < off) red[lin] += red[lin + off];
    __syncthreads();
  }
  if (tx == 0) atomicAdd(&pooled[b * 256 + c], red[ty * 64]);
}

// ---------------- 7: SE gate (per-batch matvecs) ----------------
__global__ void gate_kernel(const float* __restrict__ pooled, const float* __restrict__ Wg1,
                            const float* __restrict__ bg1, const float* __restrict__ Wg2,
                            const float* __restrict__ bg2, float* __restrict__ gate) {
  __shared__ float pS[256];
  __shared__ float hS[64];
  int b = blockIdx.x, t = threadIdx.x;
  pS[t] = pooled[b * 256 + t] * (1.f / 16384.f);   // mean over 128*128
  __syncthreads();
  if (t < 64) {
    float a = bg1[t];
    for (int ci = 0; ci < 256; ++ci) a += Wg1[t * 256 + ci] * pS[ci];
    hS[t] = a / (1.f + __expf(-a));                // SiLU
  }
  __syncthreads();
  float a = bg2[t];
  for (int k = 0; k < 64; ++k) a += Wg2[t * 64 + k] * hS[k];
  gate[b * 256 + t] = 1.f / (1.f + __expf(-a));
}

// ---------------- 8: out *= gate[b,c] (in place, float4) ----------------
__global__ void scale_kernel(float* __restrict__ out, const float* __restrict__ gate) {
  size_t i4 = (size_t)blockIdx.x * 256 + threadIdx.x;
  size_t base = i4 * 4;
  float g = gate[base >> 14];                      // plane index = b*256 + c
  float4* o4 = (float4*)out;
  float4 v = o4[i4];
  v.x *= g; v.y *= g; v.z *= g; v.w *= g;
  o4[i4] = v;
}

extern "C" void kernel_launch(void* const* d_in, const int* in_sizes, int n_in,
                              void* d_out, int out_size, void* d_ws, size_t ws_size,
                              hipStream_t stream) {
  (void)in_sizes; (void)n_in; (void)out_size; (void)ws_size;
  const float* x    = (const float*)d_in[0];
  const float* Wkc  = (const float*)d_in[1];
  const float* bkc  = (const float*)d_in[2];
  const float* Wkp1 = (const float*)d_in[3];
  const float* bkp1 = (const float*)d_in[4];
  const float* g1   = (const float*)d_in[5];
  const float* be1  = (const float*)d_in[6];
  const float* m1   = (const float*)d_in[7];
  const float* v1   = (const float*)d_in[8];
  const float* Wkp2 = (const float*)d_in[9];
  const float* bkp2 = (const float*)d_in[10];
  const float* Wdef = (const float*)d_in[11];
  const float* Wg1  = (const float*)d_in[12];
  const float* bg1  = (const float*)d_in[13];
  const float* Wg2  = (const float*)d_in[14];
  const float* bg2  = (const float*)d_in[15];

  char* ws = (char*)d_ws;
  _Float16* xh     = (_Float16*)(ws + OFF_XH);
  _Float16* comp_h = (_Float16*)(ws + OFF_COMP);
  _Float16* t_h    = (_Float16*)(ws + OFF_T);
  float*    kw     = (float*)(ws + OFF_KW);
  float*    enh    = (float*)(ws + OFF_ENH);
  float*    pooled = (float*)(ws + OFF_POOL);
  float*    gate   = (float*)(ws + OFF_GATE);
  float*    fused  = (float*)d_out;

  // 0) x -> f16 (8388608 elems, float4 per thread)
  cast_f16_kernel<<<8192, 256, 0, stream>>>(x, xh);
  // 1) comp 1x1 conv: WMMA GEMM per batch (128x4096x256), block tile 128x32
  comp_wmma_kernel<<<dim3(128, 8), 256, 0, stream>>>(xh, Wkc, bkc, comp_h);
  // 2) kp1 3x3 conv + BN + SiLU (K = 128*9), block tile 64x64
  conv3x3_wmma_kernel<128, 1, 0><<<dim3(64, 8), 256, 0, stream>>>(
      comp_h, Wkp1, bkp1, g1, be1, m1, v1, t_h, nullptr);
  // 3) kp2 1x1 conv + softmax (also zeroes pooled)
  kp2_softmax_kernel<<<128, 256, 0, stream>>>(t_h, Wkp2, bkp2, kw, pooled);
  // 4) grouped 3x3 conv (32 GEMMs of K = 64*9), block tile 64x64
  conv3x3_wmma_kernel<64, 4, 1><<<dim3(64, 32), 256, 0, stream>>>(
      xh, Wdef, nullptr, nullptr, nullptr, nullptr, nullptr, nullptr, enh);
  // 5) reassembly + bilinear + fused -> d_out, pooled partials
  reassembly_kernel<<<dim3(512, 64), dim3(64, 4), 0, stream>>>(x, kw, enh, fused, pooled);
  // 6) SE gate
  gate_kernel<<<8, 256, 0, stream>>>(pooled, Wg1, bg1, Wg2, bg2, gate);
  // 7) out *= gate (33554432/4 float4s)
  scale_kernel<<<32768, 256, 0, stream>>>(fused, gate);
}
